// Attention_32203664785681
// MI455X (gfx1250) — matrix operand order repaired and checked by reading, not compiled, not run on hardware
//
#include <hip/hip_runtime.h>

typedef __attribute__((ext_vector_type(16))) __bf16          v16bf;
typedef __attribute__((ext_vector_type(8)))  float           v8f;
typedef __attribute__((ext_vector_type(4)))  unsigned int    uint4v;
typedef __attribute__((ext_vector_type(4)))  unsigned short  ushort4v;

#define S_LEN   2048
#define DIM_C   2048
#define NH      16
#define KVH     2
#define HD      128
#define NREP    8
#define EPSF    1e-6f

union BOp { uint4v q[2]; unsigned short u[16]; v16bf v; };

__device__ __forceinline__ unsigned short f2bf(float f) {
  unsigned u = __builtin_bit_cast(unsigned, f);
  u += 0x7FFFu + ((u >> 16) & 1u);          // round-to-nearest-even
  return (unsigned short)(u >> 16);
}

// A operand (16xK tile, this lane = row l15, half = lane>>4):
// ISA 7.12.2 16-bit A 16x32: element e -> K = e + 8*half (+8 extra for e>=8)
__device__ __forceinline__ v16bf load_A_bf16(const unsigned short* __restrict__ row,
                                             int k0, int half) {
  BOp t;
  t.q[0] = *(const uint4v*)(row + k0 + 8 * half);        // e=0..7
  t.q[1] = *(const uint4v*)(row + k0 + 16 + 8 * half);   // e=8..15
  return t.v;
}

// B operand (Kx16 tile, this lane = col l15, half = lane>>4):
// B layout: same K order as A,
// element e -> K = e + 8*half (+8 extra for e>=8)
__device__ __forceinline__ v16bf load_B_bf16(const unsigned short* __restrict__ row,
                                             int k0, int half) {
  BOp t;
  t.q[0] = *(const uint4v*)(row + k0 + 8 * half);        // e=0..7
  t.q[1] = *(const uint4v*)(row + k0 + 16 + 8 * half);   // e=8..15
  return t.v;
}

__device__ __forceinline__ v8f vzero8() {
  v8f z = {0.f, 0.f, 0.f, 0.f, 0.f, 0.f, 0.f, 0.f};
  return z;
}

// ---------------------------------------------------------------------------
// Elementwise f32 -> bf16 (n divisible by 4)
// ---------------------------------------------------------------------------
__global__ __launch_bounds__(256)
void cvt_f32_bf16(const float* __restrict__ in, unsigned short* __restrict__ out,
                  int n4) {
  int t = blockIdx.x * blockDim.x + threadIdx.x;
  if (t >= n4) return;
  float4 x = *(const float4*)(in + (size_t)t * 4);
  ushort4v o = {f2bf(x.x), f2bf(x.y), f2bf(x.z), f2bf(x.w)};
  *(ushort4v*)(out + (size_t)t * 4) = o;
}

// ---------------------------------------------------------------------------
// Y[M,N] = Xb[M,K] @ Wb[N,K]^T  (bf16 in, f32 out)
// 4 waves/block; each wave computes a 16(M) x 64(N) strip: one A load feeds
// four WMMAs.
// ---------------------------------------------------------------------------
__global__ __launch_bounds__(128)
void gemm_bf16(const unsigned short* __restrict__ X,
               const unsigned short* __restrict__ W,
               float* __restrict__ Y, int M, int N, int K) {
  const int lane = threadIdx.x & 31;
  const int wave = threadIdx.x >> 5;
  const int m0 = blockIdx.y * 16;
  const int n0 = blockIdx.x * 256 + wave * 64;
  if (m0 >= M || n0 >= N) return;
  const int half = lane >> 4;
  const int l15  = lane & 15;
  const unsigned short* xrow = X + (size_t)(m0 + l15) * K;
  const unsigned short* wrow0 = W + (size_t)(n0 + l15) * K;
  v8f acc[4];
#pragma unroll
  for (int nn = 0; nn < 4; ++nn) acc[nn] = vzero8();
  for (int k0 = 0; k0 < K; k0 += 32) {
    v16bf a = load_A_bf16(xrow, k0, half);
    __builtin_prefetch(xrow + k0 + 64, 0, 3);   // WGP-scope prefetch, 2 steps ahead
#pragma unroll
    for (int nn = 0; nn < 4; ++nn) {
      const unsigned short* wrow = wrow0 + (size_t)nn * 16 * K;
      v16bf b = load_B_bf16(wrow, k0, half);
      if (nn == 0) __builtin_prefetch(wrow + k0 + 64, 0, 3);
      acc[nn] = __builtin_amdgcn_wmma_f32_16x16x32_bf16(false, a, false, b,
                                                        (short)0, acc[nn],
                                                        false, false);
    }
  }
#pragma unroll
  for (int nn = 0; nn < 4; ++nn)
#pragma unroll
    for (int r = 0; r < 8; ++r) {
      int m = r + 8 * half;
      Y[(size_t)(m0 + m) * N + n0 + nn * 16 + l15] = acc[nn][r];
    }
}

// ---------------------------------------------------------------------------
// RMSNorm + RoPE, f32 in -> bf16 out. One wave per (token, head).
// Q heads are pre-scaled by 1/sqrt(HD).
// ---------------------------------------------------------------------------
__global__ __launch_bounds__(32)
void normrope(const float* __restrict__ xqf, const float* __restrict__ xkf,
              unsigned short* __restrict__ qb, unsigned short* __restrict__ kb,
              const float* __restrict__ cosp, const float* __restrict__ sinp,
              const float* __restrict__ qg, const float* __restrict__ kg,
              int S) {
  const int tok  = blockIdx.x;
  const int h    = blockIdx.y;
  const int lane = threadIdx.x;
  const int s    = tok % S;
  const float* base; unsigned short* obase; const float* g; float oscale;
  if (h < NH) {
    base = xqf + ((size_t)tok * NH + h) * HD;
    obase = qb + ((size_t)tok * NH + h) * HD;
    g = qg; oscale = 0.08838834764831845f;        // 1/sqrt(128)
  } else {
    base = xkf + ((size_t)tok * KVH + (h - NH)) * HD;
    obase = kb + ((size_t)tok * KVH + (h - NH)) * HD;
    g = kg; oscale = 1.0f;
  }
  const int d = lane * 4;
  float4 x = *(const float4*)(base + d);
  float xa[4] = {x.x, x.y, x.z, x.w};
  float ss = xa[0]*xa[0] + xa[1]*xa[1] + xa[2]*xa[2] + xa[3]*xa[3];
#pragma unroll
  for (int off = 1; off < 32; off <<= 1) ss += __shfl_xor(ss, off, 32);
  const float inv = rsqrtf(ss * (1.0f / HD) + EPSF);
  const int d2 = (d < 64) ? d + 64 : d - 64;
  float4 xp = *(const float4*)(base + d2);
  float pa[4] = {xp.x, xp.y, xp.z, xp.w};
  const float* cb = cosp + (size_t)s * HD;
  const float* sb = sinp + (size_t)s * HD;
  const float sgn = (d < 64) ? -1.0f : 1.0f;
  ushort4v o;
#pragma unroll
  for (int i = 0; i < 4; ++i) {
    float xn = xa[i] * inv * g[d + i];
    float pn = pa[i] * inv * g[d2 + i];
    float v = (xn * cb[d + i] + sgn * pn * sb[d + i]) * oscale;
    o[i] = f2bf(v);
  }
  *(ushort4v*)(obase + d) = o;
}

// ---------------------------------------------------------------------------
// V: f32 [B,S,KVH,HD] -> bf16 transposed [B,KVH,HD,S] (keys contiguous),
// so the PV B-operand becomes contiguous b128 loads.
// ---------------------------------------------------------------------------
__global__ __launch_bounds__(256)
void v_transpose(const float* __restrict__ xvf, unsigned short* __restrict__ vt,
                 int S, int n) {
  int t = blockIdx.x * blockDim.x + threadIdx.x;
  if (t >= n) return;
  int s    = t % S;
  int rest = t / S;
  int d    = rest % HD;
  int bkv  = rest / HD;               // b*KVH + kvh
  int b = bkv / KVH, kvh = bkv % KVH;
  vt[t] = f2bf(xvf[(((size_t)b * S + s) * KVH + kvh) * HD + d]);
}

// ---------------------------------------------------------------------------
// Flash attention: one wave per (b, head, 16-query tile). 32-key steps.
// bf16 Q (pre-scaled) / K / V(transposed) in, bf16 out in [B,S,NH,HD].
// Mask: (j <= i) || (i < p && j < p).
// ---------------------------------------------------------------------------
__global__ __launch_bounds__(32)
void attn_fwd(const unsigned short* __restrict__ qb,
              const unsigned short* __restrict__ kb,
              const unsigned short* __restrict__ vt,
              unsigned short* __restrict__ out,
              const int* __restrict__ sig, int S) {
  const int qi0  = blockIdx.x * 16;
  const int h    = blockIdx.y;
  const int b    = blockIdx.z;
  const int lane = threadIdx.x & 31;
  const int half = lane >> 4;
  const int l15  = lane & 15;
  const int kvh  = h / NREP;
  const int p    = *sig;

  __shared__ unsigned short plds[16][36];   // P transpose staging (bf16)

  v16bf aq[4];
  {
    const unsigned short* qrow = qb + (((size_t)b * S + qi0 + l15) * NH + h) * HD;
#pragma unroll
    for (int c = 0; c < 4; ++c) aq[c] = load_A_bf16(qrow, 32 * c, half);
  }

  v8f o[8];
#pragma unroll
  for (int cc = 0; cc < 8; ++cc) o[cc] = vzero8();
  float mrow[8], lrow[8];
#pragma unroll
  for (int r = 0; r < 8; ++r) { mrow[r] = -3.0e38f; lrow[r] = 0.f; }

  int kmax = qi0 + 16;
  if (qi0 < p && p > kmax) kmax = p;
  const int kend = (kmax + 31) & ~31;       // multiple of 32, <= S

  for (int j0 = 0; j0 < kend; j0 += 32) {
    float sraw[2][8];
    float newm[8];
#pragma unroll
    for (int r = 0; r < 8; ++r) newm[r] = mrow[r];

    // ---- scores: two 16x16 tiles = Q(16xHD) x K^T(HDx16) ----
#pragma unroll
    for (int t = 0; t < 2; ++t) {
      const unsigned short* krow =
          kb + (((size_t)b * S + j0 + 16 * t + l15) * KVH + kvh) * HD;
      v8f sacc = vzero8();
#pragma unroll
      for (int c = 0; c < 4; ++c) {
        v16bf bk = load_B_bf16(krow, 32 * c, half);
        sacc = __builtin_amdgcn_wmma_f32_16x16x32_bf16(false, aq[c], false, bk,
                                                       (short)0, sacc, false, false);
      }
      const int j = j0 + 16 * t + l15;
#pragma unroll
      for (int r = 0; r < 8; ++r) {
        const int i = qi0 + r + 8 * half;
        float sv = sacc[r];
        bool ok = (j <= i) || (i < p && j < p);
        sv = ok ? sv : -3.0e38f;
        sraw[t][r] = sv;
        float rm = sv;
#pragma unroll
        for (int off = 1; off < 16; off <<= 1)
          rm = fmaxf(rm, __shfl_xor(rm, off, 32));
        newm[r] = fmaxf(newm[r], rm);
      }
    }

    // ---- online softmax update ----
    float fac[8];
#pragma unroll
    for (int r = 0; r < 8; ++r) {
      fac[r] = __expf(mrow[r] - newm[r]);
      lrow[r] *= fac[r];
      mrow[r] = newm[r];
    }
    float pe[2][8];
#pragma unroll
    for (int t = 0; t < 2; ++t)
#pragma unroll
      for (int r = 0; r < 8; ++r) {
        float e = __expf(sraw[t][r] - newm[r]);
        pe[t][r] = e;
        float rs = e;
#pragma unroll
        for (int off = 1; off < 16; off <<= 1) rs += __shfl_xor(rs, off, 32);
        lrow[r] += rs;
      }
#pragma unroll
    for (int cc = 0; cc < 8; ++cc)
#pragma unroll
      for (int r = 0; r < 8; ++r) o[cc][r] *= fac[r];

    // ---- transpose P (C layout -> A layout) through LDS ----
    __syncthreads();
#pragma unroll
    for (int t = 0; t < 2; ++t)
#pragma unroll
      for (int r = 0; r < 8; ++r)
        plds[r + 8 * half][t * 16 + l15] = f2bf(pe[t][r]);
    __syncthreads();
    BOp ap;
#pragma unroll
    for (int e = 0; e < 16; ++e) {
      int kk = (e < 8) ? (e + 8 * half) : (e + 8 + 8 * half);  // A mapping
      ap.u[e] = plds[l15][kk];
    }

    // ---- O += P(16x32) x V(32xHD): V from [B,KVH,HD,S] layout ----
#pragma unroll
    for (int cc = 0; cc < 8; ++cc) {
      const unsigned short* vrow =
          vt + (((size_t)b * KVH + kvh) * HD + cc * 16 + l15) * S;
      v16bf bv = load_B_bf16(vrow, j0, half);
      o[cc] = __builtin_amdgcn_wmma_f32_16x16x32_bf16(false, ap.v, false, bv,
                                                      (short)0, o[cc], false, false);
    }
    __syncthreads();
  }

  // ---- normalize and store bf16 [B,S,NH,HD] ----
#pragma unroll
  for (int cc = 0; cc < 8; ++cc)
#pragma unroll
    for (int r = 0; r < 8; ++r) {
      int m = r + 8 * half;
      out[(((size_t)b * S + qi0 + m) * NH + h) * HD + cc * 16 + l15] =
          f2bf(o[cc][r] / lrow[r]);
    }
}

// ---------------------------------------------------------------------------
extern "C" void kernel_launch(void* const* d_in, const int* in_sizes, int n_in,
                              void* d_out, int out_size, void* d_ws, size_t ws_size,
                              hipStream_t stream) {
  const float* x    = (const float*)d_in[0];
  const float* cosp = (const float*)d_in[1];
  const float* sinp = (const float*)d_in[2];
  const float* wq   = (const float*)d_in[3];
  const float* wk   = (const float*)d_in[4];
  const float* wv   = (const float*)d_in[5];
  const float* wo   = (const float*)d_in[6];
  const float* qg   = (const float*)d_in[7];
  const float* kg   = (const float*)d_in[8];
  const int*   sig  = (const int*)d_in[9];
  float* out = (float*)d_out;

  const int S = S_LEN, D = DIM_C;
  const int B = in_sizes[0] / (S * D);
  const int M = B * S;

  typedef unsigned short u16;
  u16* xb  = (u16*)d_ws;                       // M x D           bf16
  u16* wqb = xb  + (size_t)M * D;              // D x D           bf16
  u16* wkb = wqb + (size_t)D * D;              // 256 x D         bf16
  u16* wvb = wkb + (size_t)(KVH * HD) * D;     // 256 x D         bf16
  u16* wob = wvb + (size_t)(KVH * HD) * D;     // D x D           bf16
  float* xqf = (float*)(wob + (size_t)D * D);  // M x D           f32
  float* xkf = xqf + (size_t)M * D;            // M x 256         f32
  float* xvf = xkf + (size_t)M * (KVH * HD);   // M x 256         f32
  u16* qbuf = (u16*)(xvf + (size_t)M * (KVH * HD));  // M x D     bf16
  u16* kbuf = qbuf + (size_t)M * D;            // M x 256         bf16
  u16* vtb  = kbuf + (size_t)M * (KVH * HD);   // [B,KVH,HD,S]    bf16
  u16* aob  = (u16*)xqf;                       // M x D bf16, aliases dead xqf

  dim3 blk256(256), blk128(128), blk32(32);
  auto cvt = [&](const float* src, u16* dst, size_t n) {
    int n4 = (int)(n / 4);
    cvt_f32_bf16<<<dim3((n4 + 255) / 256), blk256, 0, stream>>>(src, dst, n4);
  };
  // 1) precision conversion of activations + weights (bandwidth-trivial)
  cvt(x,  xb,  (size_t)M * D);
  cvt(wq, wqb, (size_t)D * D);
  cvt(wk, wkb, (size_t)(KVH * HD) * D);
  cvt(wv, wvb, (size_t)(KVH * HD) * D);
  cvt(wo, wob, (size_t)D * D);
  // 2) QKV projections (bf16 WMMA, f32 accumulate/out)
  gemm_bf16<<<dim3(D / 256, M / 16), blk128, 0, stream>>>(xb, wqb, xqf, M, D, D);
  gemm_bf16<<<dim3(1, M / 16), blk128, 0, stream>>>(xb, wkb, xkf, M, KVH * HD, D);
  gemm_bf16<<<dim3(1, M / 16), blk128, 0, stream>>>(xb, wvb, xvf, M, KVH * HD, D);
  // 3) RMSNorm + RoPE -> bf16 Q (pre-scaled) / K
  normrope<<<dim3(M, NH + KVH), blk32, 0, stream>>>(xqf, xkf, qbuf, kbuf,
                                                    cosp, sinp, qg, kg, S);
  // 4) V -> bf16, transposed to [B,KVH,HD,S]
  {
    int n = M * KVH * HD;
    v_transpose<<<dim3((n + 255) / 256), blk256, 0, stream>>>(xvf, vtb, S, n);
  }
  // 5) flash attention -> bf16 [B,S,NH,HD] (== [M, D])
  attn_fwd<<<dim3(S / 16, NH, B), blk32, 0, stream>>>(qbuf, kbuf, vtb, aob, sig, S);
  // 6) output projection -> f32 out
  gemm_bf16<<<dim3(D / 256, M / 16), blk128, 0, stream>>>(aob, wob, out, M, D, D);
}
